// LinearAttention_90452011253812
// MI455X (gfx1250) — matrix operand order repaired and checked by reading, not compiled, not run on hardware
//
#include <hip/hip_runtime.h>
#include <math.h>

// ---- problem constants (from reference) ----
#define BATCH 16
#define CDIM  128          // channels
#define NPIX  16384        // H*W = 128*128
#define HEADS 4
#define DHEAD 32
#define SQRT_C 11.313708498984761f   // sqrt(128)
#define QSCALE 0.17677669529663688f  // 32^-0.5

typedef __attribute__((ext_vector_type(16))) __bf16 v16bf;
typedef __attribute__((ext_vector_type(8)))  __bf16 v8bf;
typedef __attribute__((ext_vector_type(8)))  float  v8f;

__device__ inline v16bf mk16(v8bf lo, v8bf hv) {
  v16bf r;
#pragma unroll
  for (int i = 0; i < 8; ++i) { r[i] = lo[i]; r[i + 8] = hv[i]; }
  return r;
}
// contiguous 16 bf16 (built from two 16B loads -> alignment-safe for LDS too)
__device__ inline v16bf ld16(const __bf16* p) {
  return mk16(*(const v8bf*)p, *(const v8bf*)(p + 8));
}
// A-fragment: row pointer, k-base; lane holds contiguous run {kb+16hi..+15}
__device__ inline v16bf lda(const __bf16* row, int kb, int hi) {
  const __bf16* p = row + kb + hi * 16;
  return mk16(*(const v8bf*)p, *(const v8bf*)(p + 8));
}
__device__ inline v8f wmma_bf16(v16bf a, v16bf b, v8f c) {
  return __builtin_amdgcn_wmma_f32_16x16x32_bf16(false, a, false, b, (short)0, c, false, false);
}

// ---- K0: zero kdenom, cast weights to bf16 ----
__global__ void la_init(const float* __restrict__ wqf, const float* __restrict__ wof,
                        __bf16* __restrict__ wq, __bf16* __restrict__ wo,
                        float* __restrict__ kden) {
  int t = blockIdx.x * 256 + threadIdx.x;
  if (t < 384 * 128) wq[t] = (__bf16)wqf[t];
  if (t < 128 * 128) wo[t] = (__bf16)wof[t];
  if (t < BATCH * HEADS * DHEAD) kden[t] = 0.f;
}

// ---- K1: channel RMS-norm + transpose to pixel-major bf16 xn[b][p][c] ----
__global__ __launch_bounds__(256) void la_rms1(const float* __restrict__ x,
                                               const float* __restrict__ g1,
                                               __bf16* __restrict__ xn) {
  int p = blockIdx.x * 256 + threadIdx.x;          // global pixel 0..262143
  size_t base = (size_t)(p >> 14) * CDIM * NPIX + (p & (NPIX - 1));
  float s = 0.f;
#pragma unroll 8
  for (int c = 0; c < CDIM; ++c) { float v = x[base + (size_t)c * NPIX]; s += v * v; }
  float inv = SQRT_C / fmaxf(sqrtf(s), 1e-12f);
  size_t ob = (size_t)p * CDIM;
  for (int c0 = 0; c0 < CDIM; c0 += 8) {
    v8bf o;
#pragma unroll
    for (int j = 0; j < 8; ++j) {
      int c = c0 + j;
      o[j] = (__bf16)(x[base + (size_t)c * NPIX] * inv * g1[c]);
    }
    *(v8bf*)(xn + ob + c0) = o;
  }
}

// ---- K2: qkv GEMM (384x128 @ 128x16 tile) + exp/softmax prep ----
__global__ __launch_bounds__(256) void la_qkv(const __bf16* __restrict__ xn,
                                              const __bf16* __restrict__ wq,
                                              __bf16* __restrict__ expq,
                                              __bf16* __restrict__ expk,
                                              __bf16* __restrict__ vmat,
                                              float* __restrict__ qden,
                                              float* __restrict__ kden) {
  __shared__ __bf16 xs[16 * 136];   // xn tile [p][c], padded stride
  __shared__ float qds[64];         // per (h,p) exp-sum
  __shared__ float kds[128];        // per k-row partial sum (one writer each)
  int b = blockIdx.x >> 10, pt = blockIdx.x & 1023, pbase = pt * 16;
  int t = threadIdx.x;
  {
    int p = t >> 4, c8 = (t & 15) * 8;
    *(v8bf*)(&xs[p * 136 + c8]) =
        *(const v8bf*)(xn + ((size_t)(b * NPIX + pbase + p)) * CDIM + c8);
  }
  if (t < 64) qds[t] = 0.f;
  __syncthreads();
  int wave = t >> 5, lane = t & 31, lane15 = lane & 15, hi = lane >> 4;
#pragma unroll
  for (int tt = 0; tt < 3; ++tt) {
    int T = wave * 3 + tt;                                   // M-tile 0..23
    const __bf16* arow = wq + (size_t)(T * 16 + lane15) * CDIM;
    v8f acc = {};
#pragma unroll
    for (int ks = 0; ks < 4; ++ks) {
      v16bf a = lda(arow, ks * 32, hi);
      v16bf bb = ld16(&xs[lane15 * 136 + ks * 32 + hi * 16]);
      acc = wmma_bf16(a, bb, acc);
    }
    int p = pbase + lane15;
    if (T < 8) {                         // ---- q rows: exp + per-pixel sum
      int h = T >> 1;
      float s = 0.f; v8bf pk;
#pragma unroll
      for (int r = 0; r < 8; ++r) { float e = __expf(acc[r]); s += e; pk[r] = (__bf16)e; }
      int d0 = (T & 1) * 16 + hi * 8;
      *(v8bf*)(expq + (((size_t)(b * HEADS + h) * NPIX + p) * DHEAD + d0)) = pk;
      atomicAdd(&qds[h * 16 + lane15], s);
    } else if (T < 16) {                 // ---- k rows: exp + row sums
      int h = (T - 8) >> 1;
#pragma unroll
      for (int r = 0; r < 8; ++r) {
        float e = __expf(acc[r]);
        int d = ((T - 8) & 1) * 16 + hi * 8 + r;
        expk[((size_t)((b * HEADS + h) * DHEAD + d)) * NPIX + p] = (__bf16)e;
        float s = e;
        s += __shfl_xor(s, 1); s += __shfl_xor(s, 2);
        s += __shfl_xor(s, 4); s += __shfl_xor(s, 8);
        if (lane15 == 0) kds[(T - 8) * 16 + hi * 8 + r] = s;
      }
    } else {                             // ---- v rows: just store bf16
      int h = (T - 16) >> 1;
#pragma unroll
      for (int r = 0; r < 8; ++r) {
        int e = ((T - 16) & 1) * 16 + hi * 8 + r;
        vmat[((size_t)((b * HEADS + h) * DHEAD + e)) * NPIX + p] = (__bf16)acc[r];
      }
    }
  }
  __syncthreads();
  if (t < 64) {
    int h = t >> 4, p = t & 15;
    qden[(size_t)(b * HEADS + h) * NPIX + pbase + p] = qds[t];
  }
  if (t < 128) atomicAdd(&kden[b * 128 + t], kds[t]);
}

// ---- K3: context = softmax_k @ v^T per (b,h), K=16384, output ctxT bf16 [e][d] ----
__global__ __launch_bounds__(256) void la_ctx(const __bf16* __restrict__ expk,
                                              const __bf16* __restrict__ vmat,
                                              const float* __restrict__ kden,
                                              __bf16* __restrict__ ctxT) {
  __shared__ float red[8 * 32 * 33];
  int bh = blockIdx.x;
  int t = threadIdx.x, wave = t >> 5, lane = t & 31, lane15 = lane & 15, hi = lane >> 4;
  const __bf16* kb_ = expk + (size_t)bh * DHEAD * NPIX;
  const __bf16* vb_ = vmat + (size_t)bh * DHEAD * NPIX;
  v8f acc[2][2] = {};
  int seg = wave * 2048;
  for (int it = 0; it < 64; ++it) {
    int kb = seg + it * 32;
    v16bf a[2], bb[2];
#pragma unroll
    for (int m = 0; m < 2; ++m)
      a[m] = lda(kb_ + (size_t)(m * 16 + lane15) * NPIX, kb, hi);
#pragma unroll
    for (int nn = 0; nn < 2; ++nn)
      bb[nn] = ld16(vb_ + (size_t)(nn * 16 + lane15) * NPIX + kb + hi * 16);
#pragma unroll
    for (int m = 0; m < 2; ++m)
#pragma unroll
      for (int nn = 0; nn < 2; ++nn) acc[m][nn] = wmma_bf16(a[m], bb[nn], acc[m][nn]);
  }
#pragma unroll
  for (int m = 0; m < 2; ++m)
#pragma unroll
    for (int nn = 0; nn < 2; ++nn)
#pragma unroll
      for (int r = 0; r < 8; ++r) {
        int d = m * 16 + hi * 8 + r, e = nn * 16 + lane15;
        red[(wave * 32 + d) * 33 + e] = acc[m][nn][r];
      }
  __syncthreads();
  for (int j = 0; j < 4; ++j) {
    int idx = t + j * 256;
    int d = idx & 31, e = idx >> 5;
    float s = 0.f;
#pragma unroll
    for (int w = 0; w < 8; ++w) s += red[(w * 32 + d) * 33 + e];
    s /= kden[bh * DHEAD + d];
    ctxT[(size_t)(bh * DHEAD + e) * DHEAD + d] = (__bf16)s;
  }
}

// ---- K4: fused attention-out + out-projection + final RMS-norm ----
__global__ __launch_bounds__(256) void la_out(const __bf16* __restrict__ ctxT,
                                              const __bf16* __restrict__ expq,
                                              const float* __restrict__ qden,
                                              const __bf16* __restrict__ wo,
                                              const float* __restrict__ b_out,
                                              const float* __restrict__ g2,
                                              float* __restrict__ out) {
  __shared__ __bf16 att[16 * 136];  // [p][c] padded
  __shared__ float nrm[16];
  int b = blockIdx.x >> 10, pt = blockIdx.x & 1023, pbase = pt * 16;
  int t = threadIdx.x, wave = t >> 5, lane = t & 31, lane15 = lane & 15, hi = lane >> 4;
  if (t < 16) nrm[t] = 0.f;
  // phase A: wave -> one 16-row c-tile of attout = ctxT @ expq (K = 32)
  {
    int h = wave >> 1, et = wave & 1;
    const __bf16* arow = ctxT + (size_t)((b * HEADS + h) * DHEAD + et * 16 + lane15) * DHEAD;
    v16bf a = lda(arow, 0, hi);
    v16bf bb = ld16(expq + ((size_t)(b * HEADS + h) * NPIX + pbase + lane15) * DHEAD + hi * 16);
    v8f acc = {};
    acc = wmma_bf16(a, bb, acc);
    float f = QSCALE / qden[(size_t)(b * HEADS + h) * NPIX + pbase + lane15];
    v8bf pk;
#pragma unroll
    for (int r = 0; r < 8; ++r) pk[r] = (__bf16)(acc[r] * f);
    *(v8bf*)(&att[lane15 * 136 + wave * 16 + hi * 8]) = pk;
  }
  __syncthreads();
  // phase B: wave -> one 16-row o-tile of w_out @ attout, then RMS norm
  {
    const __bf16* arow = wo + (size_t)(wave * 16 + lane15) * CDIM;
    v8f acc = {};
#pragma unroll
    for (int ks = 0; ks < 4; ++ks) {
      v16bf a = lda(arow, ks * 32, hi);
      v16bf bb = ld16(&att[lane15 * 136 + ks * 32 + hi * 16]);
      acc = wmma_bf16(a, bb, acc);
    }
    float y[8], s = 0.f;
#pragma unroll
    for (int r = 0; r < 8; ++r) {
      int o = wave * 16 + hi * 8 + r;
      y[r] = acc[r] + b_out[o];
      s += y[r] * y[r];
    }
    atomicAdd(&nrm[lane15], s);
    __syncthreads();
    float inv = SQRT_C / fmaxf(sqrtf(nrm[lane15]), 1e-12f);
#pragma unroll
    for (int r = 0; r < 8; ++r) {
      int o = wave * 16 + hi * 8 + r;
      out[((size_t)(b * CDIM + o)) * NPIX + pbase + lane15] = y[r] * inv * g2[o];
    }
  }
}

extern "C" void kernel_launch(void* const* d_in, const int* in_sizes, int n_in,
                              void* d_out, int out_size, void* d_ws, size_t ws_size,
                              hipStream_t stream) {
  const float* x     = (const float*)d_in[0];
  const float* g1    = (const float*)d_in[1];
  const float* w_qkv = (const float*)d_in[2];
  const float* w_out = (const float*)d_in[3];
  const float* b_out = (const float*)d_in[4];
  const float* g2    = (const float*)d_in[5];
  char* ws = (char*)d_ws;
  size_t off = 0;
  auto alloc = [&](size_t bytes) -> char* {
    char* p = ws + off; off += (bytes + 255) & ~(size_t)255; return p;
  };
  __bf16* xn   = (__bf16*)alloc((size_t)BATCH * NPIX * CDIM * 2);
  __bf16* expq = (__bf16*)alloc((size_t)BATCH * NPIX * CDIM * 2);
  __bf16* expk = (__bf16*)alloc((size_t)BATCH * NPIX * CDIM * 2);
  __bf16* vmat = (__bf16*)alloc((size_t)BATCH * NPIX * CDIM * 2);
  float*  qden = (float*) alloc((size_t)BATCH * HEADS * NPIX * 4);
  float*  kden = (float*) alloc((size_t)BATCH * HEADS * DHEAD * 4);
  __bf16* ctxT = (__bf16*)alloc((size_t)BATCH * HEADS * DHEAD * DHEAD * 2);
  __bf16* wq   = (__bf16*)alloc(384 * 128 * 2);
  __bf16* wo   = (__bf16*)alloc(128 * 128 * 2);

  la_init<<<192, 256, 0, stream>>>(w_qkv, w_out, wq, wo, kden);
  la_rms1<<<(BATCH * NPIX) / 256, 256, 0, stream>>>(x, g1, xn);
  la_qkv<<<BATCH * (NPIX / 16), 256, 0, stream>>>(xn, wq, expq, expk, vmat, qden, kden);
  la_ctx<<<BATCH * HEADS, 256, 0, stream>>>(expk, vmat, kden, ctxT);
  la_out<<<BATCH * (NPIX / 16), 256, 0, stream>>>(ctxT, expq, qden, wo, b_out, g2,
                                                  (float*)d_out);
}
